// AdditiveAttention_76484777607251
// MI455X (gfx1250) — matrix operand order repaired and checked by reading, not compiled, not run on hardware
//
#include <hip/hip_runtime.h>

#define B_    4
#define Q_    1024
#define K_    1024
#define DIN_  256
#define H_    64
#define DV_   128
#define MASK_VAL (-1000000.0f)

typedef __attribute__((ext_vector_type(16))) _Float16 v16h;
typedef __attribute__((ext_vector_type(8)))  float    v8f;

union Frag16 { uint4 u[2]; v16h h; };

// Unconditional use of the gfx1250 hardware tanh (V_TANH_F32, trans pipe).
// If this compiles, the score loop is guaranteed to run on v_tanh_f32 rather
// than the ~20-op ocml polynomial -- the decisive factor for this workload,
// which is bound by B*Q*K*H = 268M tanh evaluations.
__device__ __forceinline__ float fast_tanh(float x) {
    return __builtin_amdgcn_tanhf(x);
}

// -------------------------------------------------------------------------
// Pass 1a: q = queries @ W_q, k = keys @ W_k   (M=4096, N=64, K=256 each)
// One wave per 16x16 output tile, f16 WMMA with f32 accumulate.
// 2048 tiles total -> 256 blocks x 8 waves.
// -------------------------------------------------------------------------
__global__ __launch_bounds__(256)
void proj_kernel(const float* __restrict__ Qin, const float* __restrict__ Kin,
                 const float* __restrict__ Wq,  const float* __restrict__ Wk,
                 float* __restrict__ qh, float* __restrict__ kh) {
    const int tid  = threadIdx.x;
    const int lane = tid & 31;
    const int gw   = blockIdx.x * 8 + (tid >> 5);   // 0..2047
    const int matSel = gw >> 10;                    // 0 = q, 1 = k
    const int t  = gw & 1023;
    const int m0 = (t >> 2) << 4;                   // row tile (flattened b*1024+row)
    const int n0 = (t & 3) << 4;                    // col tile in H

    const float* src = matSel ? Kin : Qin;
    const float* W   = matSel ? Wk  : Wq;
    float*       dst = matSel ? kh  : qh;

    const int mrow  = m0 + (lane & 15);
    const int koffA = (lane >> 4) << 3;             // 0 or 8
    const int koffB = (lane >> 4) << 3;             // 0 or 8
    const int ncol  = n0 + (lane & 15);

    v8f acc = {};
    for (int kc = 0; kc < DIN_; kc += 32) {
        // A fragment: row `mrow`, two contiguous 8-float runs (16B aligned)
        const float* ap = src + (size_t)mrow * DIN_ + kc + koffA;
        float4 f0 = *(const float4*)(ap);
        float4 f1 = *(const float4*)(ap + 4);
        float4 f2 = *(const float4*)(ap + 16);
        float4 f3 = *(const float4*)(ap + 20);
        v16h A;
        A[0]=(_Float16)f0.x;  A[1]=(_Float16)f0.y;  A[2]=(_Float16)f0.z;  A[3]=(_Float16)f0.w;
        A[4]=(_Float16)f1.x;  A[5]=(_Float16)f1.y;  A[6]=(_Float16)f1.z;  A[7]=(_Float16)f1.w;
        A[8]=(_Float16)f2.x;  A[9]=(_Float16)f2.y;  A[10]=(_Float16)f2.z; A[11]=(_Float16)f2.w;
        A[12]=(_Float16)f3.x; A[13]=(_Float16)f3.y; A[14]=(_Float16)f3.z; A[15]=(_Float16)f3.w;
        // B fragment: column `ncol` of W (row-major (256,64)) -> strided reads
        const float* bp = W + (size_t)(kc + koffB) * H_ + ncol;
        v16h Bv;
        #pragma unroll
        for (int e = 0; e < 16; ++e) Bv[e] = (_Float16)bp[(e + (e & 8)) * H_];
        acc = __builtin_amdgcn_wmma_f32_16x16x32_f16(false, A, false, Bv,
                                                     (short)0, acc, false, false);
    }
    const int mbase = m0 + ((lane >> 4) << 3);
    #pragma unroll
    for (int r = 0; r < 8; ++r)
        dst[(size_t)(mbase + r) * H_ + ncol] = acc[r];
}

// -------------------------------------------------------------------------
// Pass 1b: Vt[b][n][k] = (f16) V[b][k][n]  -- column-major f16 copy of V so
// WMMA B-fragments of the PV matmul are contiguous 16B runs.
// -------------------------------------------------------------------------
__global__ __launch_bounds__(256)
void vt_kernel(const float* __restrict__ V, _Float16* __restrict__ Vt) {
    const int i = blockIdx.x * 256 + threadIdx.x;   // 0 .. 4*128*1024-1
    const int b   = i >> 17;
    const int rem = i & 131071;
    const int n   = rem >> 10;
    const int k   = rem & 1023;
    Vt[i] = (_Float16)V[((size_t)b * K_ + k) * DV_ + n];
}

// -------------------------------------------------------------------------
// Pass 2: fused additive scoring + masked softmax + PV matmul.
// One block per (batch, 16-query tile); 256 threads = 8 waves.
// LDS (dynamic, ~133KB of 320KB/WGP):
//   qh_s : 16x64 f32                    floats [0,     1024)
//   wv_s : 64 f32                       floats [1024,  1088)
//   kh_s : 128x64 f32, rows padded 65   floats [1088,  9408)
//   sc   : 16x1024 f32 scores           floats [9408,  25792)
//   pr   : 16x1032 f16 probabilities    bytes  [103168, 136192)
// -------------------------------------------------------------------------
#define KCH 128
#define PRS 1032   // padded prob row stride (halves); keeps 16B alignment

__global__ __launch_bounds__(256)
void attn_kernel(const float* __restrict__ qh, const float* __restrict__ kh,
                 const _Float16* __restrict__ Vt, const float* __restrict__ wv,
                 const int* __restrict__ vlens, float* __restrict__ out) {
    extern __shared__ float sm[];
    float*    qh_s = sm;
    float*    wv_s = sm + 1024;
    float*    kh_s = sm + 1088;
    float*    sc   = sm + 9408;
    _Float16* pr   = (_Float16*)(sm + 25792);

    const int tid  = threadIdx.x;
    const int lane = tid & 31;
    const int wave = tid >> 5;
    const int b    = blockIdx.x >> 6;
    const int m0   = (blockIdx.x & 63) << 4;
    const int vlen = vlens[b];

    // stage the 16x64 query tile + w_v
    for (int i = tid; i < 16 * H_; i += 256)
        qh_s[i] = qh[((size_t)b * Q_ + m0) * H_ + i];
    if (tid < H_) wv_s[tid] = wv[tid];
    __syncthreads();

    // ---- scores: sc[qi][k] = sum_h w_v[h] * tanh(q[qi][h] + k[k][h]) ----
    for (int kc0 = 0; kc0 < K_; kc0 += KCH) {
        for (int i = tid; i < KCH * H_; i += 256) {
            int ki = i >> 6, h = i & 63;
            kh_s[ki * 65 + h] = kh[((size_t)b * K_ + kc0 + ki) * H_ + h];
        }
        __syncthreads();
        for (int p = tid; p < 16 * KCH; p += 256) {
            int qi = p >> 7, ki = p & (KCH - 1);
            const float* qrow = qh_s + qi * H_;
            const float* krow = kh_s + ki * 65;
            float s = 0.f;
            #pragma unroll 8
            for (int h = 0; h < H_; ++h)
                s += wv_s[h] * fast_tanh(qrow[h] + krow[h]);
            int kg = kc0 + ki;
            sc[qi * K_ + kg] = (kg >= vlen) ? MASK_VAL : s;
        }
        __syncthreads();
    }

    // ---- masked softmax over full rows (exact, no online rescale) ----
    for (int row = wave * 2; row < wave * 2 + 2; ++row) {
        float m = -3.4e38f;
        for (int k = lane; k < K_; k += 32) m = fmaxf(m, sc[row * K_ + k]);
        #pragma unroll
        for (int off = 16; off > 0; off >>= 1) m = fmaxf(m, __shfl_xor(m, off, 32));
        float sum = 0.f;
        for (int k = lane; k < K_; k += 32) {
            float e = __expf(sc[row * K_ + k] - m);
            sc[row * K_ + k] = e;
            sum += e;
        }
        #pragma unroll
        for (int off = 16; off > 0; off >>= 1) sum += __shfl_xor(sum, off, 32);
        float rinv = __frcp_rn(sum);
        for (int k = lane; k < K_; k += 32)
            pr[row * PRS + k] = (_Float16)(sc[row * K_ + k] * rinv);
    }
    __syncthreads();

    // ---- O(16x128) = P(16x1024) @ V(1024x128); wave w owns 16 cols of DV ----
    const int n0    = wave << 4;
    const int koffA = (lane >> 4) << 3;   // 0 or 8
    const int koffB = (lane >> 4) << 3;   // 0 or 8
    const int arow  = lane & 15;
    const int ncol  = n0 + (lane & 15);
    const _Float16* vbase = Vt + ((size_t)b * DV_ + ncol) * K_;

    v8f acc = {};
    for (int kc = 0; kc < K_; kc += 32) {
        const _Float16* apf = pr + arow * PRS + kc + koffA;
        Frag16 A;
        A.u[0] = *(const uint4*)(apf);        // K = base..base+7
        A.u[1] = *(const uint4*)(apf + 16);   // K = base+16..base+23
        const _Float16* bpf = vbase + kc + koffB;
        Frag16 Bv;
        Bv.u[0] = *(const uint4*)(bpf);       // K = base..base+7
        Bv.u[1] = *(const uint4*)(bpf + 16);  // K = base+16..base+23
        acc = __builtin_amdgcn_wmma_f32_16x16x32_f16(false, A.h, false, Bv.h,
                                                     (short)0, acc, false, false);
    }
    const int mb = m0 + ((lane >> 4) << 3);
    #pragma unroll
    for (int r = 0; r < 8; ++r)
        out[((size_t)b * Q_ + mb + r) * DV_ + ncol] = acc[r];
}

// -------------------------------------------------------------------------
extern "C" void kernel_launch(void* const* d_in, const int* in_sizes, int n_in,
                              void* d_out, int out_size, void* d_ws, size_t ws_size,
                              hipStream_t stream) {
    const float* queries = (const float*)d_in[0];
    const float* keys    = (const float*)d_in[1];
    const float* values  = (const float*)d_in[2];
    const int*   vlens   = (const int*)  d_in[3];
    const float* Wq      = (const float*)d_in[4];
    const float* Wk      = (const float*)d_in[5];
    const float* wv      = (const float*)d_in[6];
    float*       out     = (float*)d_out;

    // workspace layout: qh (1MB f32) | kh (1MB f32) | Vt (1MB f16)  = 3MB
    float*    qh = (float*)d_ws;
    float*    kh = qh + (size_t)B_ * Q_ * H_;
    _Float16* Vt = (_Float16*)(kh + (size_t)B_ * K_ * H_);

    proj_kernel<<<256, 256, 0, stream>>>(queries, keys, Wq, Wk, qh, kh);
    vt_kernel<<<(B_ * DV_ * K_) / 256, 256, 0, stream>>>(values, Vt);

    const size_t smem = 25792 * sizeof(float) + 16 * PRS * sizeof(_Float16); // 136192 B
    attn_kernel<<<B_ * (Q_ / 16), 256, smem, stream>>>(qh, kh, Vt, wv, vlens, out);
}
